// ChargeEmbedding_44057774522738
// MI455X (gfx1250) — compile-verified, matrix operand order checked by reading, not run on hardware
//
#include <hip/hip_runtime.h>
#include <hip/hip_bf16.h>
#include <stdint.h>

// ---------------------------------------------------------------------------
// ChargeEmbedding for MI455X (gfx1250, wave32).
//
// Roofline: 3 GEMMs of [1e6 x 128] @ [128 x 128] ~= 98 GFLOP vs ~1.5 GB of
// mandatory HBM traffic (~66us @ 23.3 TB/s). bf16 WMMA (f32 accumulate) puts
// matrix math far below the memory roofline -> the pipeline is HBM-bound.
// Weights (3 x 32KB) are staged per-workgroup into LDS with the CDNA5 async
// global->LDS DMA path (ASYNCcnt + s_wait_asynccnt) so the WMMA inner loop
// feeds from ds_load_b128 instead of a serialized global-load chain.
// ---------------------------------------------------------------------------

typedef __attribute__((ext_vector_type(16))) __bf16 v16bf;
typedef __attribute__((ext_vector_type(8)))  float  v8f;
typedef int v4i __attribute__((vector_size(16)));

#define NODE_DIM 128
#define SCALE_F  0.08838834764831843f   // 1/sqrt(128)

// Async 16B global->LDS copy (tracked by ASYNCcnt).
// Builtin signature (probe-confirmed by clang diagnostic):
//   (AS1 int4* gsrc, AS3 int4* ldst, imm offset, imm cpol)
#define ASYNC_LDS_B128(gsrc, ldst)                                            \
    __builtin_amdgcn_global_load_async_to_lds_b128(                           \
        (__attribute__((address_space(1))) v4i*)(void*)(gsrc),                \
        (__attribute__((address_space(3))) v4i*)(void*)(ldst), 0, 0)

#define WAIT_ASYNC() asm volatile("s_wait_asynccnt 0x0" ::: "memory")
#define WAIT_DS()    asm volatile("s_wait_dscnt 0x0" ::: "memory")

// ---------------------------------------------------------------------------
// A-operand packer: 16-bit A 16x32 layout (ISA 7.12.2):
//   lanes 0-15 : halves 0-7 -> K=kb..kb+7,   halves 8-15 -> K=kb+16..kb+23
//   lanes 16-31: same with kb += 8
// ---------------------------------------------------------------------------
__device__ __forceinline__ v16bf packA(const float* p0, const float* p1) {
    float4 f0 = *(const float4*)(p0);
    float4 f1 = *(const float4*)(p0 + 4);
    float4 f2 = *(const float4*)(p1);
    float4 f3 = *(const float4*)(p1 + 4);
    v16bf a;
    a[0]  = (__bf16)f0.x; a[1]  = (__bf16)f0.y; a[2]  = (__bf16)f0.z; a[3]  = (__bf16)f0.w;
    a[4]  = (__bf16)f1.x; a[5]  = (__bf16)f1.y; a[6]  = (__bf16)f1.z; a[7]  = (__bf16)f1.w;
    a[8]  = (__bf16)f2.x; a[9]  = (__bf16)f2.y; a[10] = (__bf16)f2.z; a[11] = (__bf16)f2.w;
    a[12] = (__bf16)f3.x; a[13] = (__bf16)f3.y; a[14] = (__bf16)f3.z; a[15] = (__bf16)f3.w;
    return a;
}

// ---------------------------------------------------------------------------
// Kernel 0: per-graph k_g / v_g   [G,128] each, f32.
// ---------------------------------------------------------------------------
__global__ void k_kv(const float* __restrict__ charge,
                     const float* __restrict__ Wk, const float* __restrict__ Wv,
                     float* __restrict__ kg, float* __restrict__ vg, int G) {
    int idx = blockIdx.x * blockDim.x + threadIdx.x;
    if (idx >= G * NODE_DIM) return;
    int g = idx >> 7, d = idx & 127;
    float q  = charge[g];
    float qp = fmaxf(q, 0.f), qn = fmaxf(-q, 0.f);
    float np_ = qp / fmaxf(qp, 1.f);
    float nn_ = qn / fmaxf(qn, 1.f);
    kg[idx] = np_ * Wk[d] + nn_ * Wk[NODE_DIM + d];
    vg[idx] = qp  * Wv[d] + qn  * Wv[NODE_DIM + d];
}

// ---------------------------------------------------------------------------
// Weight swizzle: f32 [128(K),128(N)] row-major -> bf16 B-operand tiles.
// Tile t = kc*8+nt covers K=kc*32..+31, N=nt*16..+15; lane l holds 16 halves.
// ---------------------------------------------------------------------------
__global__ void k_swizzle_w(const float* __restrict__ Wq,
                            const float* __restrict__ W1,
                            const float* __restrict__ W2,
                            __bf16* __restrict__ Wqb,
                            __bf16* __restrict__ W1b,
                            __bf16* __restrict__ W2b) {
    int idx = blockIdx.x * blockDim.x + threadIdx.x;   // 3 * 32 tiles * 512
    if (idx >= 3 * 32 * 512) return;
    int mtx = idx / (32 * 512);
    int rem = idx - mtx * (32 * 512);
    int t = rem >> 9;            // tile 0..31
    int l = (rem >> 4) & 31;     // lane
    int h = rem & 15;            // half index
    int kc = t >> 3, nt = t & 7;
    int kb = kc * 32 + ((l < 16) ? 0 : 8);
    int K  = (h < 8) ? (kb + h) : (kb + 16 + (h - 8));
    int Nn = nt * 16 + (l & 15);
    const float* src = (mtx == 0) ? Wq : ((mtx == 1) ? W1 : W2);
    __bf16*      dst = (mtx == 0) ? Wqb : ((mtx == 1) ? W1b : W2b);
    dst[rem] = (__bf16)src[K * NODE_DIM + Nn];
}

// ---------------------------------------------------------------------------
// Kernel 1: query = ns@Wq + bq (WMMA), dot with gathered k, softplus,
// attn[n] store + atomicAdd into per-graph attn_sum.
// One wave = 16 rows. 8 waves / block. Wq staged in LDS via async DMA.
// ---------------------------------------------------------------------------
__global__ __launch_bounds__(256) void k_attn(
    const float* __restrict__ ns, const long long* __restrict__ batch,
    const __bf16* __restrict__ Wqb, const float* __restrict__ bq,
    const float* __restrict__ kg,
    float* __restrict__ attn, float* __restrict__ asum, int nTiles) {
    __shared__ __attribute__((aligned(16))) __bf16 Wlds[32 * 512];  // 32KB

    int tid = threadIdx.x;
    // Stage Wq (2048 x 16B) with async global->LDS DMA, all 256 threads.
    {
        const char* gsrc = (const char*)Wqb;
        char*       ldst = (char*)Wlds;
#pragma unroll
        for (int i = 0; i < 8; ++i) {
            int o = (i * 256 + tid) * 16;
            ASYNC_LDS_B128(gsrc + o, ldst + o);
        }
        WAIT_ASYNC();
        __syncthreads();
    }

    int wv = tid >> 5, lane = tid & 31;
    int tile = blockIdx.x * 8 + wv;
    if (tile >= nTiles) return;                 // wave-uniform, after barrier
    int base = tile * 16;
    int m = lane & 15, hf = lane >> 4;

    v8f acc[8] = {};
    const float* rowp = ns + (size_t)(base + m) * NODE_DIM;
#pragma unroll
    for (int c = 0; c < 4; ++c) {
        int kb = c * 32 + hf * 8;
        v16bf a = packA(rowp + kb, rowp + kb + 16);
#pragma unroll
        for (int nt = 0; nt < 8; ++nt) {
            v16bf b = *(const v16bf*)(Wlds + (((c << 3) | nt) * 32 + lane) * 16);
            acc[nt] = __builtin_amdgcn_wmma_f32_16x16x32_bf16(
                false, a, false, b, (short)0, acc[nt], false, false);
        }
    }
    // + bq  (C layout: lane column = nt*16 + m)
#pragma unroll
    for (int nt = 0; nt < 8; ++nt) {
        float bb = bq[nt * 16 + m];
#pragma unroll
        for (int r = 0; r < 8; ++r) acc[nt][r] += bb;
    }
    // dot(query,k) per row from the C layout; half-wave xor-shuffle reduce.
#pragma unroll
    for (int r = 0; r < 8; ++r) {
        int row = base + r + 8 * hf;
        long long g = batch[row];
        const float* kp = kg + (size_t)g * NODE_DIM + m;
        float p = 0.f;
#pragma unroll
        for (int nt = 0; nt < 8; ++nt) p += acc[nt][r] * kp[nt * 16];
        p += __shfl_xor(p, 1);
        p += __shfl_xor(p, 2);
        p += __shfl_xor(p, 4);
        p += __shfl_xor(p, 8);
        if (m == r) {
            float xs = p * SCALE_F;
            float sp = fmaxf(xs, 0.f) + __logf(1.f + __expf(-fabsf(xs)));
            attn[row] = sp;
            atomicAdd(&asum[g], sp);
        }
    }
}

// ---------------------------------------------------------------------------
// Kernel 2: x = attn*v/attn_sum -> h = silu(silu(x@W1+b1)@W2+b2)
//           out = ns + x + h
// One wave = 16 rows, 2 waves/block. One shared 32KB LDS weight buffer is
// async-staged with W1, then reused for W2 (barrier-separated). The h1 tile
// is written to LDS as bf16 directly in WMMA A-layout, so GEMM2's A operand
// is a single v16bf ds load.
// ---------------------------------------------------------------------------
__global__ __launch_bounds__(64) void k_ffn(
    const float* __restrict__ ns, const long long* __restrict__ batch,
    const float* __restrict__ vg, const float* __restrict__ attn,
    const float* __restrict__ asum,
    const __bf16* __restrict__ W1b, const float* __restrict__ b1,
    const __bf16* __restrict__ W2b, const float* __restrict__ b2,
    float* __restrict__ out, int nTiles) {
    __shared__ __attribute__((aligned(16))) __bf16 Wlds[32 * 512];      // 32KB
    __shared__ __attribute__((aligned(16))) float  Xs[2][16 * NODE_DIM]; // 16KB
    __shared__ __attribute__((aligned(16))) __bf16 Hb[2][4 * 32 * 16];   // 8KB
    __shared__ float Ss[2][16];
    __shared__ int   Gs[2][16];

    int tid = threadIdx.x;
    // Stage W1 (2048 x 16B), 64 threads x 32 chunks.
    {
        const char* gsrc = (const char*)W1b;
        char*       ldst = (char*)Wlds;
#pragma unroll 4
        for (int i = 0; i < 32; ++i) {
            int o = (i * 64 + tid) * 16;
            ASYNC_LDS_B128(gsrc + o, ldst + o);
        }
    }

    int wv = tid >> 5, lane = tid & 31;
    int tile = blockIdx.x * 2 + wv;
    if (tile >= nTiles) tile = nTiles - 1;  // clamp: duplicate work, same data
    int base = tile * 16;
    float* X = Xs[wv]; __bf16* H = Hb[wv]; float* S = Ss[wv]; int* Gg = Gs[wv];
    int m = lane & 15, hf = lane >> 4;

    if (lane < 16) {
        int row = base + lane;
        long long g = batch[row];
        S[lane]  = attn[row] / asum[g];
        Gg[lane] = (int)g;
    }
    WAIT_DS();
    // x tile into LDS (f32, linear), coalesced 128B rows
#pragma unroll 4
    for (int i = 0; i < 64; ++i) {
        int idx = i * 32 + lane;
        int r = idx >> 7, col = idx & 127;
        X[idx] = S[r] * vg[(size_t)Gg[r] * NODE_DIM + col];
    }
    WAIT_DS();
    WAIT_ASYNC();
    __syncthreads();   // W1 staged + visible

    // GEMM1: h1 = silu(x @ W1 + b1)
    v8f acc[8] = {};
#pragma unroll
    for (int c = 0; c < 4; ++c) {
        int kb = c * 32 + hf * 8;
        v16bf a = packA(&X[m * NODE_DIM + kb], &X[m * NODE_DIM + kb + 16]);
#pragma unroll
        for (int nt = 0; nt < 8; ++nt) {
            v16bf b = *(const v16bf*)(Wlds + (((c << 3) | nt) * 32 + lane) * 16);
            acc[nt] = __builtin_amdgcn_wmma_f32_16x16x32_bf16(
                false, a, false, b, (short)0, acc[nt], false, false);
        }
    }
    // silu + scatter into bf16 A-layout: element (rl, col) -> chunk c=col/32,
    // half/h per 16-bit A layout, lane = rl + 16*halfsel.
#pragma unroll
    for (int nt = 0; nt < 8; ++nt) {
        float bb = b1[nt * 16 + m];
#pragma unroll
        for (int r = 0; r < 8; ++r) {
            float hv = acc[nt][r] + bb;
            hv = hv / (1.f + __expf(-hv));               // silu
            int rl  = r + 8 * hf;
            int col = nt * 16 + m;
            int cc  = col >> 5;
            int kk  = col & 31;
            int hsel = (kk >> 3) & 1;
            int h    = (kk & 7) + ((kk >> 4) << 3);
            H[(cc * 32 + (rl + (hsel << 4))) * 16 + h] = (__bf16)hv;
        }
    }
    __syncthreads();   // everyone done reading W1 (and H written)

    // Stage W2 over the same LDS buffer.
    {
        const char* gsrc = (const char*)W2b;
        char*       ldst = (char*)Wlds;
#pragma unroll 4
        for (int i = 0; i < 32; ++i) {
            int o = (i * 64 + tid) * 16;
            ASYNC_LDS_B128(gsrc + o, ldst + o);
        }
        WAIT_ASYNC();
        __syncthreads();
    }

    // GEMM2: h2 = silu(h1 @ W2 + b2); A operand comes straight from Hb.
    v8f acc2[8] = {};
#pragma unroll
    for (int c = 0; c < 4; ++c) {
        v16bf a = *(const v16bf*)(H + (c * 32 + lane) * 16);
#pragma unroll
        for (int nt = 0; nt < 8; ++nt) {
            v16bf b = *(const v16bf*)(Wlds + (((c << 3) | nt) * 32 + lane) * 16);
            acc2[nt] = __builtin_amdgcn_wmma_f32_16x16x32_bf16(
                false, a, false, b, (short)0, acc2[nt], false, false);
        }
    }
    // Accumulate h2 into the LDS x-tile (each element owned by one lane) ...
#pragma unroll
    for (int nt = 0; nt < 8; ++nt) {
        float bb = b2[nt * 16 + m];
#pragma unroll
        for (int r = 0; r < 8; ++r) {
            float hv = acc2[nt][r] + bb;
            hv = hv / (1.f + __expf(-hv));
            int rl  = r + 8 * hf;
            int col = nt * 16 + m;
            X[rl * NODE_DIM + col] += hv;
        }
    }
    WAIT_DS();
    // ... then out = ns + (x + h2) with fully coalesced float4 traffic.
    const float* nsb  = ns  + (size_t)base * NODE_DIM;
    float*       outb = out + (size_t)base * NODE_DIM;
#pragma unroll 4
    for (int i = 0; i < 16; ++i) {
        int off = i * 128 + lane * 4;
        float4 xv = *(const float4*)&X[off];
        float4 nv = *(const float4*)&nsb[off];
        float4 ov;
        ov.x = nv.x + xv.x; ov.y = nv.y + xv.y;
        ov.z = nv.z + xv.z; ov.w = nv.w + xv.w;
        *(float4*)&outb[off] = ov;
    }
}

// ---------------------------------------------------------------------------
// Launch
// ---------------------------------------------------------------------------
extern "C" void kernel_launch(void* const* d_in, const int* in_sizes, int n_in,
                              void* d_out, int out_size, void* d_ws, size_t ws_size,
                              hipStream_t stream) {
    const float*     ns     = (const float*)d_in[0];
    const float*     charge = (const float*)d_in[1];
    const long long* batch  = (const long long*)d_in[2];
    const float*     Wq = (const float*)d_in[3];
    const float*     bq = (const float*)d_in[4];
    const float*     Wk = (const float*)d_in[5];
    const float*     Wv = (const float*)d_in[6];
    const float*     W1 = (const float*)d_in[7];
    const float*     b1 = (const float*)d_in[8];
    const float*     W2 = (const float*)d_in[9];
    const float*     b2 = (const float*)d_in[10];
    float* out = (float*)d_out;

    int N = in_sizes[0] / NODE_DIM;   // 1,000,000 (multiple of 16)
    int G = in_sizes[1];              // 10,000

    // Workspace carve-out (256B aligned slices)
    uintptr_t p = (uintptr_t)d_ws;
    auto alloc = [&](size_t bytes) -> void* {
        void* r = (void*)p;
        p = (p + bytes + 255) & ~(uintptr_t)255;
        return r;
    };
    float*  kg   = (float*)alloc((size_t)G * NODE_DIM * sizeof(float));
    float*  vg   = (float*)alloc((size_t)G * NODE_DIM * sizeof(float));
    float*  attn = (float*)alloc((size_t)N * sizeof(float));
    float*  asum = (float*)alloc((size_t)G * sizeof(float));
    __bf16* Wqb  = (__bf16*)alloc(32 * 512 * sizeof(__bf16));
    __bf16* W1b  = (__bf16*)alloc(32 * 512 * sizeof(__bf16));
    __bf16* W2b  = (__bf16*)alloc(32 * 512 * sizeof(__bf16));

    (void)hipMemsetAsync(asum, 0, (size_t)G * sizeof(float), stream);  // capture-safe

    k_kv<<<(G * NODE_DIM + 255) / 256, 256, 0, stream>>>(charge, Wk, Wv, kg, vg, G);
    k_swizzle_w<<<(3 * 32 * 512 + 255) / 256, 256, 0, stream>>>(Wq, W1, W2, Wqb, W1b, W2b);

    int nTiles = N / 16;
    k_attn<<<(nTiles + 7) / 8, 256, 0, stream>>>(ns, batch, Wqb, bq, kg, attn, asum, nTiles);
    k_ffn<<<(nTiles + 1) / 2, 64, 0, stream>>>(ns, batch, vg, attn, asum,
                                               W1b, b1, W2b, b2, out, nTiles);
}
